// MHTimeSwiGLU_67954972557423
// MI455X (gfx1250) — matrix operand order repaired and checked by reading, not compiled, not run on hardware
//
#include <hip/hip_runtime.h>
#include <hip/hip_bf16.h>
#include <cstdint>
#include <cstddef>

// ---------------------------------------------------------------------------
// MI455X (gfx1250) implementation.
//
// Key insight: no softmax => (Q K^T) V == Q (K^T V), turning three S x S
// score matrices (16 MB each, ~77 GFLOP total) into 64x64 inner matrices
// (~2.4 GFLOP). The dominant work is then the QKV projection GEMM
// (4096 x 5376 x 768 ~ 34 GFLOP), all done with v_wmma_f32_16x16x32_f16.
//
// Fragment layouts follow CDNA5 ISA 7.12.2:
//  A (16x32 f16): lane L holds row M=L%16; halves 0..7 = K kb..kb+7,
//                 halves 8..15 = K kb+16..kb+23, kb = k0 + 8*(L>=16).
//  B (32x16 f16): lane L holds col N=L%16; same K order as A,
//                 kb = k0 + 8*(L>=16).  => B operands stored as [N,K].
//  C/D (16x16 f32): lane L col N=L%16, VGPR v row M = v + 8*(L>=16).
//
// The K=2048 streaming GEMMs use CDNA5 async global->LDS copies
// (global_load_async_to_lds_b128 + s_wait_asynccnt) with double buffering,
// then ds_load_b128 fragment gathers.
// ---------------------------------------------------------------------------

#define B_  2
#define S_  2048
#define DM  768
#define H_  12
#define HD_ 64
#define N7  (7 * DM)      // 5376
#define BS  (B_ * S_)     // 4096

typedef __attribute__((ext_vector_type(16))) _Float16 v16h;
typedef __attribute__((ext_vector_type(8)))  _Float16 v8h;
typedef __attribute__((ext_vector_type(8)))  float    v8f;

union F16x16 { v16h f; v8h h[2]; };

// A fragment: row-major [M,K] f16, lda multiple of 8, k0 multiple of 32.
__device__ __forceinline__ v16h load_a_frag(const _Float16* __restrict__ A,
                                            int lda, int m0, int k0, int lane) {
  const int row = m0 + (lane & 15);
  const int kb  = k0 + ((lane >> 4) << 3);
  const _Float16* p = A + (size_t)row * lda + kb;
  F16x16 u;
  u.h[0] = *(const v8h*)(p);
  u.h[1] = *(const v8h*)(p + 16);
  return u.f;
}

// B fragment: operand stored transposed as [N,K] f16, ldb multiple of 8.
__device__ __forceinline__ v16h load_b_frag(const _Float16* __restrict__ BT,
                                            int ldb, int n0, int k0, int lane) {
  const int col = n0 + (lane & 15);
  const int kb  = k0 + ((lane >> 4) << 3);
  const _Float16* p = BT + (size_t)col * ldb + kb;
  F16x16 u;
  u.h[0] = *(const v8h*)(p);
  u.h[1] = *(const v8h*)(p + 16);
  return u.f;
}

__device__ __forceinline__ v8f wmma_f16(v16h a, v16h b, v8f c) {
  return __builtin_amdgcn_wmma_f32_16x16x32_f16(false, a, false, b,
                                                (short)0, c, false, false);
}

// --- CDNA5 async global->LDS copy (ASYNCcnt-tracked, ISA ch.10/15.18) ------
__device__ __forceinline__ void async_copy_b128(const _Float16* lds_dst,
                                                const _Float16* gsrc) {
  const uint32_t lds_off = (uint32_t)(uintptr_t)lds_dst;     // addr[31:0] = LDS offset
  const uint64_t gaddr   = (uint64_t)(uintptr_t)gsrc;
  asm volatile("global_load_async_to_lds_b128 %0, %1, off"
               :: "v"(lds_off), "v"(gaddr)
               : "memory");
}

__device__ __forceinline__ void wait_async0() {
  asm volatile("s_wait_asynccnt 0x0" ::: "memory");
}

// ---------------------------------------------------------------------------
// f32 -> f16 conversion
// ---------------------------------------------------------------------------
__global__ __launch_bounds__(256)
void f32_to_f16_kernel(const float* __restrict__ src, _Float16* __restrict__ dst, int n) {
  const int i = blockIdx.x * 256 + threadIdx.x;
  if (i < n) dst[i] = (_Float16)src[i];
}

// ---------------------------------------------------------------------------
// Generic WMMA GEMM: D[M,N] = A16[M,K] * BT16[N,K]^T + bias[N]
// Block = 256 threads = 8 waves; block tile 128(M) x 128(N);
// wave tile 32(M) x 64(N): 2 A-frags amortized over 4 B-frags -> 8 WMMA/kstep.
// M, N multiples of 128; K multiple of 32.
// ---------------------------------------------------------------------------
template <bool F16OUT>
__global__ __launch_bounds__(256)
void gemm_nk_kernel(const _Float16* __restrict__ A, const _Float16* __restrict__ BT,
                    const float* __restrict__ bias, void* __restrict__ Dp,
                    int M, int N, int K) {
  const int lane = threadIdx.x & 31;
  const int wave = threadIdx.x >> 5;
  const int m0 = blockIdx.x * 128 + (wave & 3) * 32;
  const int n0 = blockIdx.y * 128 + (wave >> 2) * 64;
  if (m0 >= M || n0 >= N) return;  // wave-uniform => EXEC all-1s at WMMA

  v8f acc[2][4] = {};
  for (int k0 = 0; k0 < K; k0 += 32) {
    const v16h a0 = load_a_frag(A, K, m0,      k0, lane);
    const v16h a1 = load_a_frag(A, K, m0 + 16, k0, lane);
    // Speculative prefetch of the next K-tile (global_prefetch_b8).
    __builtin_prefetch(A + (size_t)(m0 + (lane & 15)) * K + k0 + 32, 0, 3);
    __builtin_prefetch(A + (size_t)(m0 + 16 + (lane & 15)) * K + k0 + 32, 0, 3);
#pragma unroll
    for (int j = 0; j < 4; ++j) {
      const v16h b = load_b_frag(BT, K, n0 + 16 * j, k0, lane);
      __builtin_prefetch(BT + (size_t)(n0 + 16 * j + (lane & 15)) * K + k0 + 32, 0, 3);
      acc[0][j] = wmma_f16(a0, b, acc[0][j]);
      acc[1][j] = wmma_f16(a1, b, acc[1][j]);
    }
  }
  const int cl = lane & 15;
#pragma unroll
  for (int half = 0; half < 2; ++half) {
    const int rb = m0 + 16 * half + ((lane >> 4) << 3);
#pragma unroll
    for (int j = 0; j < 4; ++j) {
      const int col = n0 + 16 * j + cl;
      const float bv = bias ? bias[col] : 0.0f;
#pragma unroll
      for (int v = 0; v < 8; ++v) {
        const float val = acc[half][j][v] + bv;
        if (F16OUT) ((_Float16*)Dp)[(size_t)(rb + v) * N + col] = (_Float16)val;
        else        ((float*)Dp)[(size_t)(rb + v) * N + col]    = val;
      }
    }
  }
}

// ---------------------------------------------------------------------------
// Repack k1,k2,k3,v from qkv[b,s,7,h,d] into kvT[b,h,sel,d,s]  (f16)
// sel: 0=k1 (qkv idx1), 1=k2 (idx3), 2=k3 (idx5), 3=v (idx6)
// ---------------------------------------------------------------------------
__global__ __launch_bounds__(256)
void repack_kvT_kernel(const _Float16* __restrict__ qkv, _Float16* __restrict__ kvT) {
  const size_t total = (size_t)B_ * H_ * 4 * HD_ * S_;
  size_t idx = (size_t)blockIdx.x * 256 + threadIdx.x;
  if (idx >= total) return;
  const int s = (int)(idx % S_);
  size_t t = idx / S_;
  const int d = (int)(t % HD_); t /= HD_;
  const int sel = (int)(t % 4); t /= 4;
  const int h = (int)(t % H_);
  const int b = (int)(t / H_);
  const int map[4] = {1, 3, 5, 6};
  kvT[idx] = qkv[((size_t)(b * S_ + s) * 7 + map[sel]) * DM + h * HD_ + d];
}

// ---------------------------------------------------------------------------
// 64x64 "inner" GEMM over K = S: C[64,64] = Arow[64,S] * BT[64,S]^T  (f16 out)
// 512 threads = 16 waves, one 16x16 tile per wave.
// Double-buffered async global->LDS staging: per K-step, threads 0..255 move
// one 16B chunk each (A: 64x32 f16 tile, B: 64x32 f16 tile = 8 KB total),
// then all 16 waves gather fragments from LDS (ds_load_b128).
// Used as:  MjT = v^T kj  (A = vT, B = kjT)  and  M3T = u^T k3.
// ---------------------------------------------------------------------------
__device__ __forceinline__ void outer64_body(const _Float16* __restrict__ A,
                                             const _Float16* __restrict__ BT,
                                             _Float16* __restrict__ C) {
  __shared__ __align__(16) _Float16 Ab[2][64 * 32];
  __shared__ __align__(16) _Float16 Bb[2][64 * 32];

  const int tid  = threadIdx.x;
  const int lane = tid & 31;
  const int wave = tid >> 5;             // 0..15
  const int m0 = (wave & 3) * 16;
  const int n0 = (wave >> 2) * 16;

  // Stager assignment: tid 0..127 -> A tile, 128..255 -> B tile; 16B per thread.
  const int isB = (tid >> 7) & 1;
  const int row = (tid & 127) >> 1;      // 0..63
  const int chk = (tid & 1) * 8;         // half-offset of 16B chunk
  const _Float16* gbase = (isB ? BT : A) + (size_t)row * S_ + chk;

  int cur = 0;
  // Prologue: stage K-step 0 into buffer 0.
  if (tid < 256) {
    _Float16* dst = (isB ? Bb[0] : Ab[0]) + row * 32 + chk;
    async_copy_b128(dst, gbase);
  }

  v8f acc = {};
  for (int k0 = 0; k0 < S_; k0 += 32) {
    wait_async0();
    __syncthreads();                     // buffer `cur` ready; prev reads done
    if (k0 + 32 < S_ && tid < 256) {     // stage next K-step into other buffer
      _Float16* dst = (isB ? Bb[cur ^ 1] : Ab[cur ^ 1]) + row * 32 + chk;
      async_copy_b128(dst, gbase + k0 + 32);
    }
    // Fragment gathers from LDS.
    const int ar  = m0 + (lane & 15);
    const int akb = (lane >> 4) << 3;
    const _Float16* pa = Ab[cur] + ar * 32 + akb;
    F16x16 ua; ua.h[0] = *(const v8h*)pa; ua.h[1] = *(const v8h*)(pa + 16);

    const int br  = n0 + (lane & 15);
    const int bkb = (lane >> 4) << 3;
    const _Float16* pb = Bb[cur] + br * 32 + bkb;
    F16x16 ub; ub.h[0] = *(const v8h*)pb; ub.h[1] = *(const v8h*)(pb + 16);

    acc = wmma_f16(ua.f, ub.f, acc);
    cur ^= 1;
  }

  const int col = n0 + (lane & 15);
  const int rb  = m0 + ((lane >> 4) << 3);
#pragma unroll
  for (int v = 0; v < 8; ++v)
    C[(size_t)(rb + v) * HD_ + col] = (_Float16)acc[v];
}

__global__ __launch_bounds__(512)
void kv_outer_kernel(const _Float16* __restrict__ kvT, _Float16* __restrict__ MT) {
  const int bh = blockIdx.x >> 1;        // (b*H + h)
  const int j  = blockIdx.x & 1;         // 0 -> M1T, 1 -> M2T
  outer64_body(kvT + ((size_t)bh * 4 + 3) * HD_ * S_,   // vT
               kvT + ((size_t)bh * 4 + j) * HD_ * S_,   // kjT
               MT  + ((size_t)bh * 3 + j) * HD_ * HD_);
}

__global__ __launch_bounds__(512)
void u_outer_kernel(const _Float16* __restrict__ uT, const _Float16* __restrict__ kvT,
                    _Float16* __restrict__ MT) {
  const int bh = blockIdx.x;
  outer64_body(uT  + (size_t)bh * HD_ * S_,             // uT
               kvT + ((size_t)bh * 4 + 2) * HD_ * S_,   // k3T
               MT  + ((size_t)bh * 3 + 2) * HD_ * HD_);
}

// ---------------------------------------------------------------------------
// u = silu(q1 @ M1) * (q2 @ M2), stored transposed as uT[b,h,d,s] (f16).
// Grid: (B*H, S/64); 128 threads = 4 waves, each wave a 16-row strip of 64 cols.
// ---------------------------------------------------------------------------
__global__ __launch_bounds__(128)
void swiglu_t_kernel(const _Float16* __restrict__ qkv, const _Float16* __restrict__ MT,
                     _Float16* __restrict__ uT) {
  const int lane = threadIdx.x & 31;
  const int wave = threadIdx.x >> 5;
  const int bh = blockIdx.x;
  const int b = bh / H_, h = bh % H_;
  const int s0 = blockIdx.y * 64 + wave * 16;
  const int lda = 7 * DM;
  const _Float16* A1  = qkv + ((size_t)b * S_ * 7 + 0) * DM + h * HD_;  // q1
  const _Float16* A2  = qkv + ((size_t)b * S_ * 7 + 2) * DM + h * HD_;  // q2
  const _Float16* M1T = MT + ((size_t)bh * 3 + 0) * HD_ * HD_;
  const _Float16* M2T = MT + ((size_t)bh * 3 + 1) * HD_ * HD_;

  v8f acc1[4] = {};
  v8f acc2[4] = {};
#pragma unroll
  for (int k0 = 0; k0 < HD_; k0 += 32) {
    const v16h a1 = load_a_frag(A1, lda, s0, k0, lane);
    const v16h a2 = load_a_frag(A2, lda, s0, k0, lane);
#pragma unroll
    for (int j = 0; j < 4; ++j) {
      acc1[j] = wmma_f16(a1, load_b_frag(M1T, HD_, 16 * j, k0, lane), acc1[j]);
      acc2[j] = wmma_f16(a2, load_b_frag(M2T, HD_, 16 * j, k0, lane), acc2[j]);
    }
  }
  const int cl = lane & 15;
  const int rb = s0 + ((lane >> 4) << 3);
#pragma unroll
  for (int j = 0; j < 4; ++j) {
    const int n = 16 * j + cl;
#pragma unroll
    for (int v = 0; v < 8; ++v) {
      const int s = rb + v;
      const float t1 = acc1[j][v];
      const float t2 = acc2[j][v];
      const float uu = (t1 / (1.0f + __expf(-t1))) * t2;   // silu(t1) * t2
      uT[((size_t)bh * HD_ + n) * S_ + s] = (_Float16)uu;
    }
  }
}

// ---------------------------------------------------------------------------
// ctx[b,s,h*64+n] = (q3 @ M3)  (f32 out)
// ---------------------------------------------------------------------------
__global__ __launch_bounds__(128)
void ctx_kernel(const _Float16* __restrict__ qkv, const _Float16* __restrict__ MT,
                float* __restrict__ ctx) {
  const int lane = threadIdx.x & 31;
  const int wave = threadIdx.x >> 5;
  const int bh = blockIdx.x;
  const int b = bh / H_, h = bh % H_;
  const int s0 = blockIdx.y * 64 + wave * 16;
  const int lda = 7 * DM;
  const _Float16* A3  = qkv + ((size_t)b * S_ * 7 + 4) * DM + h * HD_;  // q3
  const _Float16* M3T = MT + ((size_t)bh * 3 + 2) * HD_ * HD_;

  v8f acc[4] = {};
#pragma unroll
  for (int k0 = 0; k0 < HD_; k0 += 32) {
    const v16h a = load_a_frag(A3, lda, s0, k0, lane);
#pragma unroll
    for (int j = 0; j < 4; ++j)
      acc[j] = wmma_f16(a, load_b_frag(M3T, HD_, 16 * j, k0, lane), acc[j]);
  }
  const int cl = lane & 15;
  const int rb = s0 + ((lane >> 4) << 3);
#pragma unroll
  for (int j = 0; j < 4; ++j) {
    const int n = 16 * j + cl;
#pragma unroll
    for (int v = 0; v < 8; ++v)
      ctx[(size_t)(b * S_ + rb + v) * DM + h * HD_ + n] = acc[j][v];
  }
}

// ---------------------------------------------------------------------------
// GroupNorm: per (b,h) mean/rsqrt(var) over (S, HD); then normalize -> f16.
// ---------------------------------------------------------------------------
__global__ __launch_bounds__(256)
void gn_stats_kernel(const float* __restrict__ ctx, float* __restrict__ stats) {
  __shared__ float sh0[256];
  __shared__ float sh1[256];
  const int bh = blockIdx.x, b = bh / H_, h = bh % H_;
  float s1 = 0.0f, s2 = 0.0f;
  for (int i = threadIdx.x; i < S_ * HD_; i += 256) {
    const int sr = i >> 6, c = i & 63;
    const float v = ctx[(size_t)(b * S_ + sr) * DM + h * HD_ + c];
    s1 += v; s2 += v * v;
  }
  sh0[threadIdx.x] = s1; sh1[threadIdx.x] = s2;
  __syncthreads();
  for (int off = 128; off > 0; off >>= 1) {
    if ((int)threadIdx.x < off) {
      sh0[threadIdx.x] += sh0[threadIdx.x + off];
      sh1[threadIdx.x] += sh1[threadIdx.x + off];
    }
    __syncthreads();
  }
  if (threadIdx.x == 0) {
    const float inv = 1.0f / (float)(S_ * HD_);
    const float mean = sh0[0] * inv;
    const float var  = sh1[0] * inv - mean * mean;
    stats[bh * 2 + 0] = mean;
    stats[bh * 2 + 1] = rsqrtf(var + 1e-5f);
  }
}

__global__ __launch_bounds__(256)
void gn_apply_kernel(const float* __restrict__ ctx, const float* __restrict__ stats,
                     const float* __restrict__ gw, const float* __restrict__ gb,
                     _Float16* __restrict__ ctxn) {
  const size_t idx = (size_t)blockIdx.x * 256 + threadIdx.x;
  if (idx >= (size_t)BS * DM) return;
  const int c = (int)(idx % DM);
  const int b = (int)(idx / ((size_t)S_ * DM));
  const int h = c >> 6;
  const float mean = stats[(b * H_ + h) * 2 + 0];
  const float rstd = stats[(b * H_ + h) * 2 + 1];
  ctxn[idx] = (_Float16)((ctx[idx] - mean) * rstd * gw[c] + gb[c]);
}

// ---------------------------------------------------------------------------
// Launch
// ---------------------------------------------------------------------------
extern "C" void kernel_launch(void* const* d_in, const int* in_sizes, int n_in,
                              void* d_out, int out_size, void* d_ws, size_t ws_size,
                              hipStream_t stream) {
  (void)in_sizes; (void)n_in; (void)out_size; (void)ws_size;
  const float* x  = (const float*)d_in[0];
  const float* Ww = (const float*)d_in[1];   // [5376, 768]  == [N, K]
  const float* Wb = (const float*)d_in[2];
  const float* gw = (const float*)d_in[3];
  const float* gb = (const float*)d_in[4];
  const float* ow = (const float*)d_in[5];   // [768, 768]   == [N, K]
  const float* ob = (const float*)d_in[6];

  char* ws = (char*)d_ws;
  size_t off = 0;
  auto alloc = [&](size_t bytes) -> void* {
    void* p = ws + off;
    off = (off + bytes + 255) & ~(size_t)255;
    return p;
  };

  _Float16* X16   = (_Float16*)alloc((size_t)BS * DM * 2);          //  6.3 MB
  _Float16* W16   = (_Float16*)alloc((size_t)N7 * DM * 2);          //  8.3 MB
  _Float16* OW16  = (_Float16*)alloc((size_t)DM * DM * 2);          //  1.2 MB
  _Float16* QKV16 = (_Float16*)alloc((size_t)BS * N7 * 2);          // 44.0 MB
  _Float16* KVT   = (_Float16*)alloc((size_t)B_ * H_ * 4 * HD_ * S_ * 2); // 25.2 MB
  _Float16* MT    = (_Float16*)alloc((size_t)B_ * H_ * 3 * HD_ * HD_ * 2);
  _Float16* UT    = (_Float16*)alloc((size_t)B_ * H_ * HD_ * S_ * 2);     // 6.3 MB
  float*    CTX   = (float*)   alloc((size_t)BS * DM * 4);          // 12.6 MB
  _Float16* CTXN  = (_Float16*)alloc((size_t)BS * DM * 2);          //  6.3 MB
  float*    STATS = (float*)   alloc((size_t)B_ * H_ * 2 * 4);
  float*    OUT   = (float*)d_out;

  // 1) f32 -> f16 operand conversion (everything stays resident in L2).
  f32_to_f16_kernel<<<dim3((BS * DM + 255) / 256), 256, 0, stream>>>(x, X16, BS * DM);
  f32_to_f16_kernel<<<dim3((N7 * DM + 255) / 256), 256, 0, stream>>>(Ww, W16, N7 * DM);
  f32_to_f16_kernel<<<dim3((DM * DM + 255) / 256), 256, 0, stream>>>(ow, OW16, DM * DM);

  // 2) QKV projection: [4096,5376] = X @ Wproj^T + b   (dominant GEMM, ~34 GF)
  gemm_nk_kernel<true><<<dim3(BS / 128, N7 / 128), 256, 0, stream>>>(
      X16, W16, Wb, QKV16, BS, N7, DM);

  // 3) Repack k1,k2,k3,v to [b,h,sel,d,s] for contiguous-K WMMA fragments.
  {
    const size_t total = (size_t)B_ * H_ * 4 * HD_ * S_;
    repack_kvT_kernel<<<dim3((unsigned)((total + 255) / 256)), 256, 0, stream>>>(QKV16, KVT);
  }

  // 4) M1T, M2T = v^T kj  (64x64 per head, K = 2048, async-LDS pipelined)
  kv_outer_kernel<<<dim3(B_ * H_ * 2), 512, 0, stream>>>(KVT, MT);

  // 5) u = silu(q1 M1) * (q2 M2), stored transposed
  swiglu_t_kernel<<<dim3(B_ * H_, S_ / 64), 128, 0, stream>>>(QKV16, MT, UT);

  // 6) M3T = u^T k3  (async-LDS pipelined)
  u_outer_kernel<<<dim3(B_ * H_), 512, 0, stream>>>(UT, KVT, MT);

  // 7) ctx = q3 M3  (f32)
  ctx_kernel<<<dim3(B_ * H_, S_ / 64), 128, 0, stream>>>(QKV16, MT, CTX);

  // 8) GroupNorm over (S, HD) per (b, head-group)
  gn_stats_kernel<<<dim3(B_ * H_), 256, 0, stream>>>(CTX, STATS);
  gn_apply_kernel<<<dim3((BS * DM + 255) / 256), 256, 0, stream>>>(CTX, STATS, gw, gb, CTXN);

  // 9) Output projection: [4096,768] = ctxn @ out_w^T + out_b  (f32 out)
  gemm_nk_kernel<false><<<dim3(BS / 128, DM / 128), 256, 0, stream>>>(
      CTXN, OW16, ob, OUT, BS, DM, DM);
}
